// MultiHeadAttention_14147622273066
// MI455X (gfx1250) — matrix operand order repaired and checked by reading, compiled, not run on hardware
//
#include <hip/hip_runtime.h>
#include <hip/hip_bf16.h>

typedef _Float16 h16;
typedef __attribute__((ext_vector_type(16))) _Float16 v16h;
typedef __attribute__((ext_vector_type(8)))  _Float16 v8h;
typedef __attribute__((ext_vector_type(8)))  float    v8f;
typedef __attribute__((ext_vector_type(4)))  int      v4i;

typedef __attribute__((address_space(1))) v4i gas_v4i;
typedef __attribute__((address_space(3))) v4i las_v4i;

union V16 { v16h v; v8h h[2]; };

#define NB     2
#define TSEQ   2048
#define DMODEL 2048
#define NH     16
#define DK     128
#define WIN    256

#define WMMA_F16(A, B, C) \
  __builtin_amdgcn_wmma_f32_16x16x32_f16(false, (A), false, (B), (short)0, (C), false, false)

#if __has_builtin(__builtin_amdgcn_global_load_async_to_lds_b128) && \
    __has_builtin(__builtin_amdgcn_s_wait_asynccnt)
#define HAS_ASYNC 1
#else
#define HAS_ASYNC 0
#endif

// stage one 16B chunk global -> LDS (async path if available)
__device__ __forceinline__ void stage16(const h16* g, h16* l) {
#if HAS_ASYNC
  __builtin_amdgcn_global_load_async_to_lds_b128(
      (gas_v4i*)(h16*)g, (las_v4i*)l, 0, 0);
#else
  *(v8h*)l = *(const v8h*)g;
#endif
}

__device__ __forceinline__ void wait_async() {
#if HAS_ASYNC
  __builtin_amdgcn_s_wait_asynccnt(0);
#endif
}

// ---------------------------------------------------------------- convert
__global__ void cvt_f32_f16(const float* __restrict__ in, h16* __restrict__ out, int n) {
  int i = blockIdx.x * blockDim.x + threadIdx.x;
  if (i < n) out[i] = (h16)in[i];
}

// ---------------------------------------------------------------- GEMM core
// block = 128 threads (4 waves); block tile 128(M) x 64(N); K-step 32,
// A/B tiles double-buffered in LDS via async global->LDS; each wave: 32x64.
__device__ __forceinline__ void stage_tiles(const h16* __restrict__ A,
                                            const h16* __restrict__ Bw,
                                            h16* Ab, h16* Bb,
                                            int gm0, int n0, int k0, int tid) {
#pragma unroll
  for (int j = 0; j < 4; ++j) {               // A: 128 rows x 32 halfs = 512 chunks
    const int c = tid + j * 128;
    const int row = c >> 2, part = c & 3;
    stage16(A + (size_t)(gm0 + row) * DMODEL + k0 + part * 8, Ab + row * 32 + part * 8);
  }
#pragma unroll
  for (int j = 0; j < 2; ++j) {               // B: 64 rows x 32 halfs = 256 chunks
    const int c = tid + j * 128;
    const int row = c >> 2, part = c & 3;
    stage16(Bw + (size_t)(n0 + row) * DMODEL + k0 + part * 8, Bb + row * 32 + part * 8);
  }
}

__device__ __forceinline__ void gemm_tile(const h16* __restrict__ A,
                                          const h16* __restrict__ Bw,
                                          v8f acc[2][4], int gm0, int n0) {
  __shared__ __align__(16) h16 Ab[2][128 * 32];
  __shared__ __align__(16) h16 Bb[2][64 * 32];

  const int tid  = threadIdx.x;
  const int w    = tid >> 5;
  const int lane = tid & 31;
  const int ln   = lane & 15;
  const bool hi  = lane >= 16;
  const int kb   = hi ? 16 : 0;

  int buf = 0;
  stage_tiles(A, Bw, Ab[0], Bb[0], gm0, n0, 0, tid);

  for (int kt = 0; kt < DMODEL / 32; ++kt) {
    wait_async();
    __syncthreads();                       // staged tile visible; prev reads done
    if (kt + 1 < DMODEL / 32)
      stage_tiles(A, Bw, Ab[buf ^ 1], Bb[buf ^ 1], gm0, n0, (kt + 1) * 32, tid);

    const h16* ab = Ab[buf];
    const h16* bb = Bb[buf];
    V16 a0, a1;
    {
      const h16* p0 = ab + (w * 32 + ln) * 32;
      const h16* p1 = ab + (w * 32 + 16 + ln) * 32;
      a0.h[0] = *(const v8h*)(p0 + kb);  a0.h[1] = *(const v8h*)(p0 + kb + 8);
      a1.h[0] = *(const v8h*)(p1 + kb);  a1.h[1] = *(const v8h*)(p1 + kb + 8);
    }
#pragma unroll
    for (int nt = 0; nt < 4; ++nt) {
      v16h bf = *(const v16h*)(bb + (nt * 16 + ln) * 32 + (hi ? 16 : 0));
      acc[0][nt] = WMMA_F16(a0.v, bf, acc[0][nt]);
      acc[1][nt] = WMMA_F16(a1.v, bf, acc[1][nt]);
    }
    buf ^= 1;
  }
}

// ---------------------------------------------------------------- QKV GEMM
__global__ __launch_bounds__(128) void gemm_qkv(
    const h16* __restrict__ xh,
    const h16* __restrict__ Wq, const h16* __restrict__ Wk, const h16* __restrict__ Wv,
    h16* __restrict__ Qh, h16* __restrict__ Kh, h16* __restrict__ Vt) {
  const int w    = threadIdx.x >> 5;
  const int lane = threadIdx.x & 31;
  const int ln   = lane & 15;
  const bool hi  = lane >= 16;

  const int gm0 = blockIdx.y * 128;
  const int n0  = blockIdx.x * 64;
  const h16* W  = (blockIdx.z == 0) ? Wq : (blockIdx.z == 1) ? Wk : Wv;

  v8f acc[2][4] = {};
  gemm_tile(xh, W, acc, gm0, n0);

#pragma unroll
  for (int mi = 0; mi < 2; ++mi) {
#pragma unroll
    for (int nt = 0; nt < 4; ++nt) {
#pragma unroll
      for (int r = 0; r < 8; ++r) {
        const int gm = gm0 + w * 32 + mi * 16 + r + (hi ? 8 : 0);  // row in [B*T]
        const int j  = n0 + nt * 16 + ln;                          // feature in [D]
        const int b = gm >> 11, t = gm & (TSEQ - 1);
        const int h = j >> 7,  d = j & (DK - 1);
        const h16 val = (h16)acc[mi][nt][r];
        if (blockIdx.z == 0)
          Qh[(((size_t)(b * NH + h)) * TSEQ + t) * DK + d] = val;
        else if (blockIdx.z == 1)
          Kh[(((size_t)(b * NH + h)) * TSEQ + t) * DK + d] = val;
        else
          Vt[(((size_t)(b * NH + h)) * DK + d) * TSEQ + t] = val;  // V transposed
      }
    }
  }
}

// ---------------------------------------------------------------- RoPE (in place on Q,K)
__global__ void rope_qk(h16* __restrict__ Qh, h16* __restrict__ Kh) {
  const int idx = blockIdx.x * blockDim.x + threadIdx.x;   // over B*H * T * 64
  if (idx >= NB * NH * TSEQ * 64) return;
  const int i  = idx & 63;
  const int t  = (idx >> 6) & (TSEQ - 1);
  const int bh = idx >> 17;
  const float inv = __expf(-(2.0f * (float)i / 128.0f) * 9.210340371976184f); // ln(1e4)
  const float ang = (float)t * inv;
  const float c = __cosf(ang), s = __sinf(ang);
  const size_t base = ((size_t)bh * TSEQ + t) * DK;
  {
    float a = (float)Qh[base + i], b = (float)Qh[base + i + 64];
    Qh[base + i]      = (h16)(a * c - b * s);
    Qh[base + i + 64] = (h16)(b * c + a * s);
  }
  {
    float a = (float)Kh[base + i], b = (float)Kh[base + i + 64];
    Kh[base + i]      = (h16)(a * c - b * s);
    Kh[base + i + 64] = (h16)(b * c + a * s);
  }
}

// ---------------------------------------------------------------- sliding-window attention
__global__ __launch_bounds__(128) void attn_swa(
    const h16* __restrict__ Qh, const h16* __restrict__ Kh,
    const h16* __restrict__ Vt, h16* __restrict__ Oh) {
  __shared__ h16 Plds[4][16 * 40];

  const int w    = threadIdx.x >> 5;
  const int lane = threadIdx.x & 31;
  const int ln   = lane & 15;
  const bool hi  = lane >= 16;
  const int kb   = hi ? 16 : 0;

  const int qb = blockIdx.x * 4 + w;
  const int q0 = qb * 16;
  const int bh = blockIdx.y;
  const int b  = bh >> 4, h = bh & 15;

  const h16* Qb = Qh + (size_t)bh * TSEQ * DK;
  const h16* Kb = Kh + (size_t)bh * TSEQ * DK;
  const h16* Vb = Vt + (size_t)bh * DK * TSEQ;

  V16 qf[4];
  {
    const h16* qp = Qb + (size_t)(q0 + ln) * DK;
#pragma unroll
    for (int dt = 0; dt < 4; ++dt) {
      qf[dt].h[0] = *(const v8h*)(qp + dt * 32 + kb);
      qf[dt].h[1] = *(const v8h*)(qp + dt * 32 + kb + 8);
    }
  }

  v8f O[8] = {};
  float m[8], l[8];
#pragma unroll
  for (int r = 0; r < 8; ++r) { m[r] = -1e30f; l[r] = 0.0f; }
  const float scale = 0.08838834764831845f;  // 1/sqrt(128)

  int kstart = q0 - WIN + 1;
  if (kstart < 0) kstart = 0;
  kstart &= ~31;
  h16* Pw = &Plds[w][0];

  for (int k0 = kstart; k0 <= q0 + 15; k0 += 32) {
    const int key0 = k0 + ln;
    const int key1 = k0 + 16 + ln;
    const int kr0 = (key0 < TSEQ) ? key0 : TSEQ - 1;
    const int kr1 = (key1 < TSEQ) ? key1 : TSEQ - 1;
    const int doff = hi ? 16 : 0;

    v8f S0 = {}, S1 = {};
#pragma unroll
    for (int dt = 0; dt < 4; ++dt) {
      v16h b0 = *(const v16h*)(Kb + (size_t)kr0 * DK + dt * 32 + doff);
      S0 = WMMA_F16(qf[dt].v, b0, S0);
    }
#pragma unroll
    for (int dt = 0; dt < 4; ++dt) {
      v16h b1 = *(const v16h*)(Kb + (size_t)kr1 * DK + dt * 32 + doff);
      S1 = WMMA_F16(qf[dt].v, b1, S1);
    }

    const int qbase = q0 + (hi ? 8 : 0);
#pragma unroll
    for (int r = 0; r < 8; ++r) {
      const int q = qbase + r;
      float s0 = S0[r] * scale, s1 = S1[r] * scale;
      const int d0 = q - key0, d1 = q - key1;
      const bool v0 = (d0 >= 0) && (d0 < WIN);
      const bool v1 = (d1 >= 0) && (d1 < WIN);
      s0 = v0 ? s0 : -1e30f;
      s1 = v1 ? s1 : -1e30f;

      float tm = fmaxf(s0, s1);
#pragma unroll
      for (int msk = 1; msk < 16; msk <<= 1) tm = fmaxf(tm, __shfl_xor(tm, msk, 32));
      const float mn = fmaxf(m[r], tm);
      const float rsc = __expf(m[r] - mn);
      m[r] = mn;
      l[r] *= rsc;
#pragma unroll
      for (int nt = 0; nt < 8; ++nt) O[nt][r] *= rsc;

      const float p0 = v0 ? __expf(s0 - mn) : 0.0f;
      const float p1 = v1 ? __expf(s1 - mn) : 0.0f;
      float rs = p0 + p1;
#pragma unroll
      for (int msk = 1; msk < 16; msk <<= 1) rs += __shfl_xor(rs, msk, 32);
      l[r] += rs;

      const int prow = r + (hi ? 8 : 0);
      Pw[prow * 40 + ln]      = (h16)p0;
      Pw[prow * 40 + 16 + ln] = (h16)p1;
    }

    V16 pf;
    {
      const h16* pp = Pw + ln * 40;
      pf.h[0] = *(const v8h*)(pp + kb);
      pf.h[1] = *(const v8h*)(pp + kb + 8);
    }

    const int kkb = k0 + (hi ? 16 : 0);
#pragma unroll
    for (int nt = 0; nt < 8; ++nt) {
      const int d = nt * 16 + ln;
      v16h vf = *(const v16h*)(Vb + (size_t)d * TSEQ + kkb);
      O[nt] = WMMA_F16(pf.v, vf, O[nt]);
    }
  }

#pragma unroll
  for (int r = 0; r < 8; ++r) {
    const float inv = 1.0f / l[r];
    const int t = q0 + r + (hi ? 8 : 0);
#pragma unroll
    for (int nt = 0; nt < 8; ++nt) {
      Oh[((size_t)b * TSEQ + t) * DMODEL + h * DK + nt * 16 + ln] = (h16)(O[nt][r] * inv);
    }
  }
}

// ---------------------------------------------------------------- output projection -> f32
__global__ __launch_bounds__(128) void gemm_out(
    const h16* __restrict__ Oh, const h16* __restrict__ Wo, float* __restrict__ out) {
  const int w    = threadIdx.x >> 5;
  const int lane = threadIdx.x & 31;
  const int ln   = lane & 15;
  const bool hi  = lane >= 16;

  const int gm0 = blockIdx.y * 128;
  const int n0  = blockIdx.x * 64;

  v8f acc[2][4] = {};
  gemm_tile(Oh, Wo, acc, gm0, n0);

#pragma unroll
  for (int mi = 0; mi < 2; ++mi) {
#pragma unroll
    for (int nt = 0; nt < 4; ++nt) {
#pragma unroll
      for (int r = 0; r < 8; ++r) {
        const int gm = gm0 + w * 32 + mi * 16 + r + (hi ? 8 : 0);
        const int j  = n0 + nt * 16 + ln;
        out[(size_t)gm * DMODEL + j] = acc[mi][nt][r];
      }
    }
  }
}

// ---------------------------------------------------------------- launcher
extern "C" void kernel_launch(void* const* d_in, const int* in_sizes, int n_in,
                              void* d_out, int out_size, void* d_ws, size_t ws_size,
                              hipStream_t stream) {
  (void)in_sizes; (void)n_in; (void)out_size; (void)ws_size;
  const float* x  = (const float*)d_in[0];
  const float* Wq = (const float*)d_in[1];
  const float* Wk = (const float*)d_in[2];
  const float* Wv = (const float*)d_in[3];
  const float* Wo = (const float*)d_in[4];

  char* ws = (char*)d_ws;
  const size_t SZ_X  = (size_t)NB * TSEQ * DMODEL * sizeof(h16);   // 16 MB
  const size_t SZ_W  = (size_t)DMODEL * DMODEL * sizeof(h16);      //  8 MB
  const size_t SZ_H  = (size_t)NB * NH * TSEQ * DK * sizeof(h16);  // 16 MB

  h16* xh  = (h16*)(ws);
  h16* Wqh = (h16*)(ws + SZ_X);
  h16* Wkh = (h16*)(ws + SZ_X + SZ_W);
  h16* Wvh = (h16*)(ws + SZ_X + 2 * SZ_W);
  h16* Woh = (h16*)(ws + SZ_X + 3 * SZ_W);
  h16* Qh  = (h16*)(ws + SZ_X + 4 * SZ_W);
  h16* Kh  = (h16*)(ws + SZ_X + 4 * SZ_W + SZ_H);
  h16* Vt  = (h16*)(ws + SZ_X + 4 * SZ_W + 2 * SZ_H);
  h16* Oh  = (h16*)(ws + SZ_X + 4 * SZ_W + 3 * SZ_H + 256);  // slack after Vt

  const int nX = NB * TSEQ * DMODEL;      // 8388608
  const int nW = DMODEL * DMODEL;         // 4194304
  cvt_f32_f16<<<(nX + 255) / 256, 256, 0, stream>>>(x,  xh,  nX);
  cvt_f32_f16<<<(nW + 255) / 256, 256, 0, stream>>>(Wq, Wqh, nW);
  cvt_f32_f16<<<(nW + 255) / 256, 256, 0, stream>>>(Wk, Wkh, nW);
  cvt_f32_f16<<<(nW + 255) / 256, 256, 0, stream>>>(Wv, Wvh, nW);
  cvt_f32_f16<<<(nW + 255) / 256, 256, 0, stream>>>(Wo, Woh, nW);

  gemm_qkv<<<dim3(DMODEL / 64, (NB * TSEQ) / 128, 3), 128, 0, stream>>>(
      xh, Wqh, Wkh, Wvh, Qh, Kh, Vt);

  const int nR = NB * NH * TSEQ * 64;     // 4194304
  rope_qk<<<(nR + 255) / 256, 256, 0, stream>>>(Qh, Kh);

  attn_swa<<<dim3((TSEQ / 16) / 4, NB * NH), 128, 0, stream>>>(Qh, Kh, Vt, Oh);

  gemm_out<<<dim3(DMODEL / 64, (NB * TSEQ) / 128), 128, 0, stream>>>(Oh, Woh, (float*)d_out);
}
